// SelfAttention_70755291235095
// MI455X (gfx1250) — matrix operand order repaired and checked by reading, not compiled, not run on hardware
//
#include <hip/hip_runtime.h>
#include <hip/hip_bf16.h>
#include <cstdint>
#include <cstddef>

#define EMBED 1024
#define HEADS 16
#define HDIM  64
#define NB    2
#define SEQ   2048

typedef __attribute__((ext_vector_type(16))) __bf16 v16bf;
typedef __attribute__((ext_vector_type(2)))  __bf16 v2bf;
typedef __attribute__((ext_vector_type(8)))  float  v8f;
typedef __attribute__((ext_vector_type(4)))  unsigned int u32x4;
typedef __attribute__((ext_vector_type(4)))  float  f32x4;

#if defined(__has_builtin)
#if __has_builtin(__builtin_amdgcn_cvt_pk_bf16_f32)
#define HAVE_PK_BF16 1
#endif
#endif

union Frag {
    u32x4 q[2];
    unsigned int u[8];
    v16bf v;
};

__device__ __forceinline__ v8f v8f_zero() {
    v8f z = {0.f, 0.f, 0.f, 0.f, 0.f, 0.f, 0.f, 0.f};
    return z;
}

__device__ __forceinline__ unsigned int bf16_rne(float x) {
    unsigned int u = __float_as_uint(x);
    u += 0x7FFFu + ((u >> 16) & 1u);   // round-to-nearest-even
    return u >> 16;
}

// packed f32x2 -> bf16x2 (low = a, high = b); hardware v_cvt_pk_bf16_f32 if available
__device__ __forceinline__ unsigned int pk_bf16(float a, float b) {
#ifdef HAVE_PK_BF16
    v2bf r = __builtin_amdgcn_cvt_pk_bf16_f32(a, b);
    return __builtin_bit_cast(unsigned int, r);
#else
    return bf16_rne(a) | (bf16_rne(b) << 16);
#endif
}

__device__ __forceinline__ unsigned short bf16s(float x) {
#ifdef HAVE_PK_BF16
    return (unsigned short)(pk_bf16(x, x) & 0xffffu);
#else
    return (unsigned short)bf16_rne(x);
#endif
}

// D = A*B + C, 16x16x32 bf16 -> f32
__device__ __forceinline__ v8f wmma_bf(const Frag& a, const Frag& b, v8f c) {
    return __builtin_amdgcn_wmma_f32_16x16x32_bf16(
        false, a.v, false, b.v, (short)0, c, false, false);
}

// A fragment 16x32 from bf16 row-major [row][k], leading dim ld.
// Lanes 0-15 row M=lane; vgpr0-3 = K (8*half..+7), vgpr4-7 = K+16.
__device__ __forceinline__ void load_a_bf(Frag& f, const unsigned short* base,
                                          int ld, int r0, int k0, int lane) {
    const int half = lane >> 4, m = lane & 15;
    const unsigned short* p = base + (size_t)(r0 + m) * ld + k0 + 8 * half;
    f.q[0] = *reinterpret_cast<const u32x4*>(p);
    f.q[1] = *reinterpret_cast<const u32x4*>(p + 16);
}

// B fragment 32x16 loaded from the TRANSPOSED source (row-major N x K, ld = ldK).
// Per lane: col N = lane&15; vgpr0-3 = K (8*half..+7), vgpr4-7 = K+16.
__device__ __forceinline__ void load_bt_bf(Frag& f, const unsigned short* bt,
                                           int ld, int n0, int k0, int lane) {
    const int half = lane >> 4, n = lane & 15;
    const unsigned short* p = bt + (size_t)(n0 + n) * ld + k0 + 8 * half;
    f.q[0] = *reinterpret_cast<const u32x4*>(p);
    f.q[1] = *reinterpret_cast<const u32x4*>(p + 16);
}

// A fragment from fp32 row-major source, converting to bf16 in-register.
__device__ __forceinline__ void load_a_f32cvt(Frag& f, const float* base,
                                              int ld, int r0, int k0, int lane) {
    const int half = lane >> 4, m = lane & 15;
    const float* p = base + (size_t)(r0 + m) * ld + k0;
    f32x4 x0 = *reinterpret_cast<const f32x4*>(p + 8 * half);
    f32x4 x1 = *reinterpret_cast<const f32x4*>(p + 8 * half + 4);
    f32x4 x2 = *reinterpret_cast<const f32x4*>(p + 16 + 8 * half);
    f32x4 x3 = *reinterpret_cast<const f32x4*>(p + 16 + 8 * half + 4);
    f.u[0] = pk_bf16(x0.x, x0.y); f.u[1] = pk_bf16(x0.z, x0.w);
    f.u[2] = pk_bf16(x1.x, x1.y); f.u[3] = pk_bf16(x1.z, x1.w);
    f.u[4] = pk_bf16(x2.x, x2.y); f.u[5] = pk_bf16(x2.z, x2.w);
    f.u[6] = pk_bf16(x3.x, x3.y); f.u[7] = pk_bf16(x3.z, x3.w);
}

// ---------------- fp32 -> bf16 convert (weights) ----------------
__global__ void cvt_bf16_kernel(const float* __restrict__ in,
                                unsigned short* __restrict__ out, int n) {
    int i = blockIdx.x * blockDim.x + threadIdx.x;
    if (i < n) out[i] = bf16s(in[i]);
}

// ---------------- per-head QKV projection: y = x @ W.T ----------------
// One wave per (n, h, s-tile). Output bf16: TR==0 -> [n][h][s][d];
// TR==1 -> [n][h][d][s] (for V, so the PV B-operand later loads contiguously).
template <int TR>
__global__ void __launch_bounds__(256)
proj_kernel(const float* __restrict__ in, const unsigned short* __restrict__ Wbf,
            unsigned short* __restrict__ out) {
    const int lane = threadIdx.x & 31;
    const int wave = (blockIdx.x * blockDim.x + threadIdx.x) >> 5;
    const int st = wave & (SEQ / 16 - 1);
    const int nh = wave >> 7;              // SEQ/16 == 128
    const int h = nh & (HEADS - 1);
    const int n = nh >> 4;

    const float* abase = in + (size_t)n * SEQ * EMBED + h * HDIM;
    Frag a0, a1;
    load_a_f32cvt(a0, abase, EMBED, st * 16, 0, lane);
    load_a_f32cvt(a1, abase, EMBED, st * 16, 32, lane);

    const int col = lane & 15, half = lane >> 4;
#pragma unroll
    for (int ct = 0; ct < 4; ++ct) {
        Frag b0, b1;
        load_bt_bf(b0, Wbf, HDIM, ct * 16, 0, lane);   // W row-major == B^T form
        load_bt_bf(b1, Wbf, HDIM, ct * 16, 32, lane);
        v8f c = v8f_zero();
        c = wmma_bf(a0, b0, c);
        c = wmma_bf(a1, b1, c);
        if (TR == 0) {
            unsigned short* o = out + ((size_t)(n * HEADS + h) * SEQ + st * 16) * HDIM + ct * 16;
#pragma unroll
            for (int r = 0; r < 8; ++r)
                o[(size_t)(r + 8 * half) * HDIM + col] = bf16s(c[r]);
        } else {
            unsigned short* o = out + ((size_t)(n * HEADS + h) * HDIM + ct * 16 + col) * SEQ
                                + st * 16 + 8 * half;
#pragma unroll
            for (int r = 0; r < 8; ++r)
                o[r] = bf16s(c[r]);
        }
    }
}

// ---------------- flash attention per (n, h, q-tile of 16) ----------------
__global__ void __launch_bounds__(256)
flash_kernel(const unsigned short* __restrict__ Qp, const unsigned short* __restrict__ Kp,
             const unsigned short* __restrict__ VpT, unsigned short* __restrict__ O) {
    __shared__ unsigned short Pl[8][16 * 32];   // per-wave 16x32 P tile (bf16)
    const int lane = threadIdx.x & 31;
    const int wid = threadIdx.x >> 5;
    const int wave = (blockIdx.x * blockDim.x + threadIdx.x) >> 5;
    const int qt = wave & (SEQ / 16 - 1);
    const int nh = wave >> 7;
    const int h = nh & (HEADS - 1);
    const int n = nh >> 4;
    unsigned short* pl = Pl[wid];

    const unsigned short* Qb = Qp + (size_t)(n * HEADS + h) * SEQ * HDIM;
    const unsigned short* Kb = Kp + (size_t)(n * HEADS + h) * SEQ * HDIM;
    const unsigned short* Vb = VpT + (size_t)(n * HEADS + h) * HDIM * SEQ;

    Frag aq0, aq1;
    load_a_bf(aq0, Qb, HDIM, qt * 16, 0, lane);
    load_a_bf(aq1, Qb, HDIM, qt * 16, 32, lane);

    float m[8], l[8];
    v8f acc[4];
#pragma unroll
    for (int r = 0; r < 8; ++r) { m[r] = -1e30f; l[r] = 0.f; }
#pragma unroll
    for (int t = 0; t < 4; ++t) acc[t] = v8f_zero();

    const int col = lane & 15, half = lane >> 4;
    const float scale = 0.03125f;   // 1/sqrt(EMBED) = 1/32

    for (int kb = 0; kb < SEQ; kb += 32) {
        Frag b;
        v8f s0 = v8f_zero(), s1 = v8f_zero();
        load_bt_bf(b, Kb, HDIM, kb, 0, lane);       s0 = wmma_bf(aq0, b, s0);
        load_bt_bf(b, Kb, HDIM, kb, 32, lane);      s0 = wmma_bf(aq1, b, s0);
        load_bt_bf(b, Kb, HDIM, kb + 16, 0, lane);  s1 = wmma_bf(aq0, b, s1);
        load_bt_bf(b, Kb, HDIM, kb + 16, 32, lane); s1 = wmma_bf(aq1, b, s1);

        // prefetch next K tile (per-lane addresses cover all 32 rows -> whole tile)
        if (kb + 32 < SEQ) {
            const unsigned short* nk0 =
                Kb + (size_t)(kb + 32 + col) * HDIM + 16 * half;
            const unsigned short* nk1 =
                Kb + (size_t)(kb + 48 + col) * HDIM + 16 * half;
            __builtin_prefetch(nk0, 0, 1);
            __builtin_prefetch(nk1, 0, 1);
        }

        float fc[8];
#pragma unroll
        for (int r = 0; r < 8; ++r) {
            float e0 = s0[r] * scale, e1 = s1[r] * scale;
            float rmax = fmaxf(e0, e1);
#pragma unroll
            for (int msk = 1; msk < 16; msk <<= 1)
                rmax = fmaxf(rmax, __shfl_xor(rmax, msk, 32));
            float mn = fmaxf(m[r], rmax);
            float p0 = __expf(e0 - mn), p1 = __expf(e1 - mn);
            float rs = p0 + p1;
#pragma unroll
            for (int msk = 1; msk < 16; msk <<= 1)
                rs += __shfl_xor(rs, msk, 32);
            float corr = __expf(m[r] - mn);
            l[r] = l[r] * corr + rs;
            m[r] = mn;
            fc[r] = corr;
            // write P tile row-major [q][l] (row = r+8*half, cols lane&15 and +16)
            pl[(r + 8 * half) * 32 + col] = bf16s(p0);
            pl[(r + 8 * half) * 32 + 16 + col] = bf16s(p1);
        }
#pragma unroll
        for (int t = 0; t < 4; ++t)
#pragma unroll
            for (int r = 0; r < 8; ++r)
                acc[t][r] *= fc[r];

        Frag ap;
        load_a_bf(ap, pl, 32, 0, 0, lane);          // P as A fragment from LDS
#pragma unroll
        for (int t = 0; t < 4; ++t) {
            Frag bv;
            load_bt_bf(bv, Vb, SEQ, t * 16, kb, lane);  // V^T is B^T form
            acc[t] = wmma_bf(ap, bv, acc[t]);
        }
    }

    unsigned short* ob = O + ((size_t)n * SEQ + qt * 16) * EMBED + h * HDIM;
#pragma unroll
    for (int t = 0; t < 4; ++t)
#pragma unroll
        for (int r = 0; r < 8; ++r) {
            float val = acc[t][r] / l[r];
            ob[(size_t)(r + 8 * half) * EMBED + t * 16 + col] = bf16s(val);
        }
}

// ---------------- output projection: out = O @ Wo.T + bo (fp32 out) ----------------
__global__ void __launch_bounds__(256)
outproj_kernel(const unsigned short* __restrict__ O, const unsigned short* __restrict__ Wobf,
               const float* __restrict__ bo, float* __restrict__ out) {
    const int lane = threadIdx.x & 31;
    const int wave = (blockIdx.x * blockDim.x + threadIdx.x) >> 5;
    const int ct = wave & 63;          // EMBED/16 col tiles
    const int rt = wave >> 6;          // (N*S)/16 = 256 row tiles
    v8f c = v8f_zero();
    for (int k0 = 0; k0 < EMBED; k0 += 32) {
        Frag a, b;
        load_a_bf(a, O, EMBED, rt * 16, k0, lane);
        load_bt_bf(b, Wobf, EMBED, ct * 16, k0, lane);
        c = wmma_bf(a, b, c);
    }
    const int col = lane & 15, half = lane >> 4;
    float bias = bo[ct * 16 + col];
#pragma unroll
    for (int r = 0; r < 8; ++r)
        out[(size_t)(rt * 16 + r + 8 * half) * EMBED + ct * 16 + col] = c[r] + bias;
}

extern "C" void kernel_launch(void* const* d_in, const int* in_sizes, int n_in,
                              void* d_out, int out_size, void* d_ws, size_t ws_size,
                              hipStream_t stream) {
    const float* q  = (const float*)d_in[0];
    const float* k  = (const float*)d_in[1];
    const float* v  = (const float*)d_in[2];
    const float* Wq = (const float*)d_in[3];
    const float* Wk = (const float*)d_in[4];
    const float* Wv = (const float*)d_in[5];
    const float* Wo = (const float*)d_in[6];
    const float* bo = (const float*)d_in[7];

    char* ws = (char*)d_ws;
    unsigned short* wq_bf = (unsigned short*)(ws + 0);          //   8 KB
    unsigned short* wk_bf = (unsigned short*)(ws + 8192);       //   8 KB
    unsigned short* wv_bf = (unsigned short*)(ws + 16384);      //   8 KB
    unsigned short* wo_bf = (unsigned short*)(ws + 24576);      //   2 MB
    unsigned short* Qp    = (unsigned short*)(ws + 2121728);    //   8 MB [n][h][s][d]
    unsigned short* Kp    = (unsigned short*)(ws + 10510336);   //   8 MB [n][h][s][d]
    unsigned short* VpT   = (unsigned short*)(ws + 18898944);   //   8 MB [n][h][d][s]
    unsigned short* Oat   = (unsigned short*)(ws + 27287552);   //   8 MB [n][s][e]

    cvt_bf16_kernel<<<16, 256, 0, stream>>>(Wq, wq_bf, HDIM * HDIM);
    cvt_bf16_kernel<<<16, 256, 0, stream>>>(Wk, wk_bf, HDIM * HDIM);
    cvt_bf16_kernel<<<16, 256, 0, stream>>>(Wv, wv_bf, HDIM * HDIM);
    cvt_bf16_kernel<<<4096, 256, 0, stream>>>(Wo, wo_bf, EMBED * EMBED);

    // 4096 waves each: (N*H*S/16)
    proj_kernel<0><<<512, 256, 0, stream>>>(q, wq_bf, Qp);
    proj_kernel<0><<<512, 256, 0, stream>>>(k, wk_bf, Kp);
    proj_kernel<1><<<512, 256, 0, stream>>>(v, wv_bf, VpT);

    flash_kernel<<<512, 256, 0, stream>>>(Qp, Kp, VpT, Oat);

    // 16384 waves: 256 row-tiles x 64 col-tiles
    outproj_kernel<<<2048, 256, 0, stream>>>(Oat, wo_bf, bo, (float*)d_out);
}
